// WireframeDetector_292057776219
// MI455X (gfx1250) — compile-verified, matrix operand order checked by reading, not run on hardware
//
#include <hip/hip_runtime.h>
#include <hip/hip_bf16.h>

#define HW 16384
#define Wd 128
#define Hd 128
#define LN 16384
#define BN 2
#define TOPKN 300

typedef __attribute__((ext_vector_type(16))) __bf16   v16bf;
typedef __attribute__((ext_vector_type(8)))  float    v8f;
typedef __attribute__((ext_vector_type(4)))  float    fv4;
typedef __attribute__((ext_vector_type(4)))  unsigned uv4;
typedef __attribute__((ext_vector_type(2)))  unsigned uv2;

__device__ __forceinline__ unsigned short f2bf(float f) {
  unsigned u = __builtin_bit_cast(unsigned, f);
  unsigned r = u + 0x7fffu + ((u >> 16) & 1u);   // round-to-nearest-even
  return (unsigned short)(r >> 16);
}
__device__ __forceinline__ float bf2f(unsigned short h) {
  unsigned u = ((unsigned)h) << 16;
  return __builtin_bit_cast(float, u);
}

// ---------------------------------------------------------------------------
// Generic bf16 WMMA GEMM: Out[M,N] = act(A[M,K] * B[K,N] + bias)
// AMODE 0: A is bf16 row-major [M][K] (lda in elems)
// AMODE 1: A is f32 column-major source: A[m][k] = src[k*lda + m]
// BMODE 0: B is bf16 row-major [K][N] (ldb in elems)
// BMODE 1: B is f32 with B[k][n] = src[n*ldb + k] (i.e. src row-major [N][K])
// Tile: 256 (M) x 64 (N) per block of 256 threads (8 waves), K-step 32,
// double-buffered LDS; each wave register-blocks 2x4 WMMA tiles (32x64).
// ---------------------------------------------------------------------------
template <int AMODE, int BMODE, int RELU, int OUTF32>
__global__ __launch_bounds__(256) void wmma_gemm_kernel(
    const void* __restrict__ Ap, const void* __restrict__ Bp,
    const float* __restrict__ bias, void* __restrict__ Op,
    int K, int lda, int ldb, int ldc, size_t aStride, size_t oStride) {
  __shared__ unsigned short ldsA[2][256 * 32];  // [buf][m][k]  16KB each
  __shared__ unsigned short ldsB[2][64 * 32];   // [buf][n][k]   4KB each

  const int z = blockIdx.z;
  const int m0 = blockIdx.x * 256;
  const int n0 = blockIdx.y * 64;
  const int tid = threadIdx.x;
  const int w = tid >> 5;
  const int tl = tid & 31;
  const int half = tl >> 4;
  const int mn = tl & 15;

  v8f acc[2][4];
#pragma unroll
  for (int mf = 0; mf < 2; ++mf)
#pragma unroll
    for (int nt = 0; nt < 4; ++nt)
#pragma unroll
      for (int r = 0; r < 8; ++r) acc[mf][nt][r] = 0.0f;

  // ---- stage A tile (bf16 [m][k], 256x32) into LDS buffer ----
  auto stageA = [&](int buf, int k0) {
    if (AMODE == 0) {
      const unsigned short* A = (const unsigned short*)Ap + z * aStride;
#pragma unroll
      for (int u = 0; u < 4; ++u) {
        int lin = tid + u * 256;          // 1024 units of 16B
        int row = lin >> 2;
        int q = (lin & 3) * 8;
        *reinterpret_cast<uv4*>(&ldsA[buf][row * 32 + q]) =
            *reinterpret_cast<const uv4*>(A + (size_t)(m0 + row) * lda + k0 + q);
      }
      if (k0 + 32 < K)
        __builtin_prefetch(
            (const unsigned short*)Ap + z * aStride +
                (size_t)(m0 + (tid >> 1)) * lda + k0 + 32,
            0, 1);
    } else {
      const float* A = (const float*)Ap + z * aStride;
      int k = tid >> 3;                    // 0..31
      int mb = (tid & 7) * 32;             // 0..224
      const fv4* src =
          reinterpret_cast<const fv4*>(A + (size_t)(k0 + k) * lda + m0 + mb);
#pragma unroll
      for (int v4 = 0; v4 < 8; ++v4) {
        fv4 f = src[v4];
        ldsA[buf][(mb + v4 * 4 + 0) * 32 + k] = f2bf(f.x);
        ldsA[buf][(mb + v4 * 4 + 1) * 32 + k] = f2bf(f.y);
        ldsA[buf][(mb + v4 * 4 + 2) * 32 + k] = f2bf(f.z);
        ldsA[buf][(mb + v4 * 4 + 3) * 32 + k] = f2bf(f.w);
      }
    }
  };
  // ---- stage B tile transposed to [n][k] (64x32) into LDS buffer ----
  auto stageB = [&](int buf, int k0) {
    if (BMODE == 0) {
      const unsigned short* Bm = (const unsigned short*)Bp;
      int k = tid >> 3;                    // 0..31
      int nb = (tid & 7) * 8;              // 0..56
      uv4 uq = *reinterpret_cast<const uv4*>(Bm + (size_t)(k0 + k) * ldb + n0 + nb);
#pragma unroll
      for (int i = 0; i < 4; ++i) {
        unsigned v = uq[i];
        ldsB[buf][(nb + i * 2 + 0) * 32 + k] = (unsigned short)v;
        ldsB[buf][(nb + i * 2 + 1) * 32 + k] = (unsigned short)(v >> 16);
      }
    } else {
      const float* Bm = (const float*)Bp;
      int n = tid >> 2;                    // 0..63
      int q = (tid & 3) * 8;               // 0..24
      const fv4* src =
          reinterpret_cast<const fv4*>(Bm + (size_t)(n0 + n) * ldb + k0 + q);
      fv4 f0 = src[0], f1 = src[1];
      uv4 d;
      d.x = (unsigned)f2bf(f0.x) | ((unsigned)f2bf(f0.y) << 16);
      d.y = (unsigned)f2bf(f0.z) | ((unsigned)f2bf(f0.w) << 16);
      d.z = (unsigned)f2bf(f1.x) | ((unsigned)f2bf(f1.y) << 16);
      d.w = (unsigned)f2bf(f1.z) | ((unsigned)f2bf(f1.w) << 16);
      *reinterpret_cast<uv4*>(&ldsB[buf][n * 32 + q]) = d;
    }
  };

  union Frag { v16bf v; uv4 q[2]; };

  // ---- compute 8 WMMAs from one LDS buffer (2 A-frags x 4 B-frags) ----
  auto compute = [&](int buf) {
    Frag a0, a1;
    int abase = (w * 32 + mn) * 32 + half * 8;   // ISA 16-bit A 16x32 layout
    a0.q[0] = *reinterpret_cast<const uv4*>(&ldsA[buf][abase]);
    a0.q[1] = *reinterpret_cast<const uv4*>(&ldsA[buf][abase + 16]);
    a1.q[0] = *reinterpret_cast<const uv4*>(&ldsA[buf][abase + 16 * 32]);
    a1.q[1] = *reinterpret_cast<const uv4*>(&ldsA[buf][abase + 16 * 32 + 16]);
#pragma unroll
    for (int nt = 0; nt < 4; ++nt) {
      Frag bfrag;
      int bbase = (nt * 16 + mn) * 32 + half * 8;
      bfrag.q[0] = *reinterpret_cast<const uv4*>(&ldsB[buf][bbase]);
      bfrag.q[1] = *reinterpret_cast<const uv4*>(&ldsB[buf][bbase + 16]);
      acc[0][nt] = __builtin_amdgcn_wmma_f32_16x16x32_bf16(
          false, a0.v, false, bfrag.v, (short)0, acc[0][nt], false, false);
      acc[1][nt] = __builtin_amdgcn_wmma_f32_16x16x32_bf16(
          false, a1.v, false, bfrag.v, (short)0, acc[1][nt], false, false);
    }
  };

  const int nk = K / 32;
  stageA(0, 0);
  stageB(0, 0);
  __syncthreads();
  for (int kt = 0; kt < nk; ++kt) {
    int cur = kt & 1;
    if (kt + 1 < nk) {                 // stage next tile into other buffer,
      stageA(cur ^ 1, (kt + 1) * 32);  // overlapping with WMMA below
      stageB(cur ^ 1, (kt + 1) * 32);
    }
    compute(cur);
    __syncthreads();
  }

  // ---- epilogue: D VGPR r -> row r + 8*half, col = lane&15 ----
#pragma unroll
  for (int mf = 0; mf < 2; ++mf) {
#pragma unroll
    for (int nt = 0; nt < 4; ++nt) {
      int col = n0 + nt * 16 + mn;
      float bv = bias[col];
#pragma unroll
      for (int r = 0; r < 8; ++r) {
        int row = m0 + w * 32 + mf * 16 + r + half * 8;
        float v = acc[mf][nt][r] + bv;
        if (RELU) v = fmaxf(v, 0.0f);
        if (OUTF32)
          ((float*)Op + z * oStride)[(size_t)row * ldc + col] = v;
        else
          ((unsigned short*)Op + z * oStride)[(size_t)row * ldc + col] = f2bf(v);
      }
    }
  }
}

// ---------------------------------------------------------------------------
__global__ void cvt_bf16_kernel(const float* __restrict__ src,
                                unsigned short* __restrict__ dst, int n4) {
  int i = blockIdx.x * 256 + threadIdx.x;
  if (i >= n4) return;
  fv4 f = reinterpret_cast<const fv4*>(src)[i];
  uv2 d;
  d.x = (unsigned)f2bf(f.x) | ((unsigned)f2bf(f.y) << 16);
  d.y = (unsigned)f2bf(f.z) | ((unsigned)f2bf(f.w) << 16);
  reinterpret_cast<uv2*>(dst)[i] = d;
}

// softmax(ch5,ch6)[1] then 3x3 NMS (neighbors recomputed -> bitwise identical)
__global__ void jloc_nms_kernel(const float* __restrict__ out9,
                                float* __restrict__ jloc) {
  int b = blockIdx.y;
  int p = blockIdx.x * 256 + threadIdx.x;
  const float* o5 = out9 + ((size_t)b * 9 + 5) * HW;
  const float* o6 = o5 + HW;
  int x = p & (Wd - 1), y = p >> 7;
  float v = 1.0f / (1.0f + expf(o5[p] - o6[p]));
  float mx = v;
  for (int dy = -1; dy <= 1; ++dy)
    for (int dx = -1; dx <= 1; ++dx) {
      int yy = y + dy, xx = x + dx;
      if (yy >= 0 && yy < Hd && xx >= 0 && xx < Wd) {
        int q = yy * Wd + xx;
        mx = fmaxf(mx, 1.0f / (1.0f + expf(o5[q] - o6[q])));
      }
    }
  jloc[(size_t)b * HW + p] = (v == mx) ? v : 0.0f;
}

// exact top-k by rank (stable tie-break by index == lax.top_k semantics)
__global__ __launch_bounds__(256) void topk_kernel(const float* __restrict__ jloc,
                                                   float* __restrict__ tks,
                                                   int* __restrict__ tki) {
  int b = blockIdx.y;
  int i = blockIdx.x * 256 + threadIdx.x;
  const float* a = jloc + (size_t)b * HW;
  float vi = a[i];
  __shared__ float tile[256];
  int rank = 0;
  for (int jt = 0; jt < 64; ++jt) {
    tile[threadIdx.x] = a[jt * 256 + threadIdx.x];
    __syncthreads();
#pragma unroll 8
    for (int jj = 0; jj < 256; ++jj) {
      int j = jt * 256 + jj;
      float vj = tile[jj];
      rank += (vj > vi) || (vj == vi && j < i);
    }
    __syncthreads();
  }
  if (rank < TOPKN) {
    tks[b * TOPKN + rank] = vi;
    tki[b * TOPKN + rank] = i;
  }
}

__global__ void junction_kernel(const float* __restrict__ out9,
                                const float* __restrict__ tks,
                                const int* __restrict__ tki,
                                float* __restrict__ juncs) {
  int b = blockIdx.y;
  int t = threadIdx.x;
  if (t >= TOPKN) return;
  const float* o7 = out9 + ((size_t)b * 9 + 7) * HW;
  const float* o8 = o7 + HW;
  int idx = tki[b * TOPKN + t];
  float s = tks[b * TOPKN + t];
  float jx = (float)(idx & (Wd - 1)) + 1.0f / (1.0f + expf(-o7[idx]));
  float jy = (float)(idx >> 7) + 1.0f / (1.0f + expf(-o8[idx]));
  if (!(s > 0.008f)) { jx = 1.0e6f; jy = 1.0e6f; }
  juncs[(b * TOPKN + t) * 2 + 0] = jx;
  juncs[(b * TOPKN + t) * 2 + 1] = jy;
}

__global__ __launch_bounds__(128) void match_kernel(
    const float* __restrict__ lp, const float* __restrict__ juncs,
    float* __restrict__ lines4, int* __restrict__ keep,
    float* __restrict__ outLines) {
  __shared__ float jx[TOPKN], jy[TOPKN];
  int b = blockIdx.y;
  for (int u = threadIdx.x; u < TOPKN; u += 128) {
    jx[u] = juncs[(b * TOPKN + u) * 2 + 0];
    jy[u] = juncs[(b * TOPKN + u) * 2 + 1];
  }
  __syncthreads();
  int l = blockIdx.x * 128 + threadIdx.x;
  size_t base = ((size_t)b * LN + l) * 4;
  float x1 = lp[base], y1 = lp[base + 1], x2 = lp[base + 2], y2 = lp[base + 3];
  float bd1 = 3.4e38f, bd2 = 3.4e38f;
  int i1 = 0, i2 = 0;
  for (int j = 0; j < TOPKN; ++j) {
    float dx = x1 - jx[j], dy = y1 - jy[j];
    float d = dx * dx + dy * dy;
    if (d < bd1) { bd1 = d; i1 = j; }
    dx = x2 - jx[j]; dy = y2 - jy[j];
    d = dx * dx + dy * dy;
    if (d < bd2) { bd2 = d; i2 = j; }
  }
  int imin = (i1 < i2) ? i1 : i2;
  int imax = (i1 < i2) ? i2 : i1;
  float px1 = jx[imin], py1 = jy[imin], px2 = jx[imax], py2 = jy[imax];
  lines4[base] = px1; lines4[base + 1] = py1;
  lines4[base + 2] = px2; lines4[base + 3] = py2;
  outLines[base]     = px1 * 4.0f;   // IMG_W / W = 4
  outLines[base + 1] = py1 * 4.0f;
  outLines[base + 2] = px2 * 4.0f;
  outLines[base + 3] = py2 * 4.0f;
  keep[(size_t)b * LN + l] = (imin < imax) ? 1 : 0;
}

// bilinear sample 32 pts per line over loi [hw][128], max-pool groups of 4,
// emit xp bf16 [L][c*8+g]
__global__ __launch_bounds__(128) void sample_kernel(
    const float* __restrict__ lines4, const float* __restrict__ loi,
    unsigned short* __restrict__ xp) {
  int b = blockIdx.y, l = blockIdx.x, c = threadIdx.x;
  size_t base = ((size_t)b * LN + l) * 4;
  float x1 = lines4[base], y1 = lines4[base + 1];
  float x2 = lines4[base + 2], y2 = lines4[base + 3];
  const float* F = loi + (size_t)b * HW * 128;
  unsigned short* xo = xp + ((size_t)b * LN + l) * 1024;
#pragma unroll
  for (int g = 0; g < 8; ++g) {
    float mx = -3.4e38f;
#pragma unroll
    for (int p = 0; p < 4; ++p) {
      float t = (float)(g * 4 + p) * (1.0f / 31.0f);
      float px = x1 * t + x2 * (1.0f - t) - 0.5f;
      float py = y1 * t + y2 * (1.0f - t) - 0.5f;
      float fx0 = fminf(fmaxf(floorf(px), 0.0f), 127.0f);
      float fy0 = fminf(fmaxf(floorf(py), 0.0f), 127.0f);
      float fx1 = fminf(fx0 + 1.0f, 127.0f);
      float fy1 = fminf(fy0 + 1.0f, 127.0f);
      int ix0 = (int)fx0, iy0 = (int)fy0, ix1 = (int)fx1, iy1 = (int)fy1;
      float w00 = (fy1 - py) * (fx1 - px);
      float w10 = (py - fy0) * (fx1 - px);
      float w01 = (fy1 - py) * (px - fx0);
      float w11 = (py - fy0) * (px - fx0);
      float v = F[((size_t)iy0 * Wd + ix0) * 128 + c] * w00 +
                F[((size_t)iy1 * Wd + ix0) * 128 + c] * w10 +
                F[((size_t)iy0 * Wd + ix1) * 128 + c] * w01 +
                F[((size_t)iy1 * Wd + ix1) * 128 + c] * w11;
      mx = fmaxf(mx, v);
    }
    xo[c * 8 + g] = f2bf(mx);
  }
}

// final 1024-dot + sigmoid + keep; one wave32 per line
__global__ __launch_bounds__(256) void head_kernel(
    const unsigned short* __restrict__ h2, const float* __restrict__ w3,
    const float* __restrict__ b3, const int* __restrict__ keep,
    float* __restrict__ scoresOut, float* __restrict__ keepOut) {
  int b = blockIdx.y;
  int w = threadIdx.x >> 5, tl = threadIdx.x & 31;
  int l = blockIdx.x * 8 + w;
  const unsigned short* hr = h2 + ((size_t)b * LN + l) * 1024 + tl * 32;
  const float* wr = w3 + tl * 32;
  const uv4* hq = reinterpret_cast<const uv4*>(hr);
  const fv4* wq = reinterpret_cast<const fv4*>(wr);
  float s = 0.0f;
#pragma unroll
  for (int i = 0; i < 4; ++i) {
    uv4 hv = hq[i];
    fv4 wa = wq[i * 2], wb = wq[i * 2 + 1];
    unsigned v;
    v = hv.x; s += bf2f((unsigned short)v) * wa.x + bf2f((unsigned short)(v >> 16)) * wa.y;
    v = hv.y; s += bf2f((unsigned short)v) * wa.z + bf2f((unsigned short)(v >> 16)) * wa.w;
    v = hv.z; s += bf2f((unsigned short)v) * wb.x + bf2f((unsigned short)(v >> 16)) * wb.y;
    v = hv.w; s += bf2f((unsigned short)v) * wb.z + bf2f((unsigned short)(v >> 16)) * wb.w;
  }
#pragma unroll
  for (int off = 16; off > 0; off >>= 1) s += __shfl_xor(s, off, 32);
  if (tl == 0) {
    float logit = s + b3[0];
    float sc = 1.0f / (1.0f + expf(-logit));
    size_t o = (size_t)b * LN + l;
    scoresOut[o] = sc;
    keepOut[o] = (keep[o] != 0 && sc > 0.05f) ? 1.0f : 0.0f;
  }
}

// ---------------------------------------------------------------------------
extern "C" void kernel_launch(void* const* d_in, const int* in_sizes, int n_in,
                              void* d_out, int out_size, void* d_ws,
                              size_t ws_size, hipStream_t stream) {
  (void)in_sizes; (void)n_in; (void)out_size; (void)ws_size;
  const float* output     = (const float*)d_in[0];
  const float* features   = (const float*)d_in[1];
  const float* line_preds = (const float*)d_in[2];
  const float* conv_w     = (const float*)d_in[3];
  const float* conv_b     = (const float*)d_in[4];
  const float* w1         = (const float*)d_in[5];
  const float* b1         = (const float*)d_in[6];
  const float* w2         = (const float*)d_in[7];
  const float* b2         = (const float*)d_in[8];
  const float* w3         = (const float*)d_in[9];
  const float* b3         = (const float*)d_in[10];

  auto al = [](size_t x) { return (x + 255) & ~(size_t)255; };
  size_t cur = 0;
  size_t off_loi   = cur; cur = al(cur + (size_t)BN * HW * 128 * 4);
  size_t off_jloc  = cur; cur = al(cur + (size_t)BN * HW * 4);
  size_t off_tks   = cur; cur = al(cur + (size_t)BN * TOPKN * 4);
  size_t off_tki   = cur; cur = al(cur + (size_t)BN * TOPKN * 4);
  size_t off_jun   = cur; cur = al(cur + (size_t)BN * TOPKN * 2 * 4);
  size_t off_lines = cur; cur = al(cur + (size_t)BN * LN * 4 * 4);
  size_t off_keep  = cur; cur = al(cur + (size_t)BN * LN * 4);
  size_t off_xp    = cur; cur = al(cur + (size_t)BN * LN * 1024 * 2);  // also h2
  size_t off_h1    = cur; cur = al(cur + (size_t)BN * LN * 1024 * 2);
  size_t off_w1b   = cur; cur = al(cur + (size_t)1024 * 1024 * 2);
  size_t off_w2b   = cur; cur = al(cur + (size_t)1024 * 1024 * 2);

  char* ws = (char*)d_ws;
  float* loi            = (float*)(ws + off_loi);
  float* jloc           = (float*)(ws + off_jloc);
  float* tks            = (float*)(ws + off_tks);
  int*   tki            = (int*)(ws + off_tki);
  float* juncs          = (float*)(ws + off_jun);
  float* lines4         = (float*)(ws + off_lines);
  int*   keep           = (int*)(ws + off_keep);
  unsigned short* xp    = (unsigned short*)(ws + off_xp);   // reused as h2
  unsigned short* h1    = (unsigned short*)(ws + off_h1);
  unsigned short* w1b   = (unsigned short*)(ws + off_w1b);
  unsigned short* w2b   = (unsigned short*)(ws + off_w2b);

  float* outLines  = (float*)d_out;
  float* outScores = outLines + (size_t)BN * LN * 4;
  float* outKeep   = outScores + (size_t)BN * LN;

  // weights -> bf16
  cvt_bf16_kernel<<<1024, 256, 0, stream>>>(w1, w1b, 1024 * 1024 / 4);
  cvt_bf16_kernel<<<1024, 256, 0, stream>>>(w2, w2b, 1024 * 1024 / 4);

  // 1x1 conv as GEMM: loi[pix][128] = features^T * conv_w^T + conv_b
  wmma_gemm_kernel<1, 1, 0, 1><<<dim3(HW / 256, 128 / 64, BN), 256, 0, stream>>>(
      features, conv_w, conv_b, loi, /*K=*/256, /*lda=*/HW, /*ldb=*/256,
      /*ldc=*/128, (size_t)256 * HW, (size_t)HW * 128);

  jloc_nms_kernel<<<dim3(HW / 256, BN), 256, 0, stream>>>(output, jloc);
  topk_kernel<<<dim3(HW / 256, BN), 256, 0, stream>>>(jloc, tks, tki);
  junction_kernel<<<dim3(1, BN), 320, 0, stream>>>(output, tks, tki, juncs);
  match_kernel<<<dim3(LN / 128, BN), 128, 0, stream>>>(line_preds, juncs, lines4,
                                                       keep, outLines);
  sample_kernel<<<dim3(LN, BN), 128, 0, stream>>>(lines4, loi, xp);

  // FC1: h1 = relu(xp * w1 + b1)   (bf16 in/out, f32 accumulate)
  wmma_gemm_kernel<0, 0, 1, 0><<<dim3(LN / 256, 1024 / 64, BN), 256, 0, stream>>>(
      xp, w1b, b1, h1, /*K=*/1024, 1024, 1024, 1024, (size_t)LN * 1024,
      (size_t)LN * 1024);
  // FC2: h2 = relu(h1 * w2 + b2)  -> reuse xp buffer
  wmma_gemm_kernel<0, 0, 1, 0><<<dim3(LN / 256, 1024 / 64, BN), 256, 0, stream>>>(
      h1, w2b, b2, xp, /*K=*/1024, 1024, 1024, 1024, (size_t)LN * 1024,
      (size_t)LN * 1024);

  head_kernel<<<dim3(LN / 8, BN), 256, 0, stream>>>(xp, w3, b3, keep, outScores,
                                                    outKeep);
}
